// CustomGRU_1511828488536
// MI455X (gfx1250) — matrix operand order repaired and checked by reading, not compiled, not run on hardware
//
#include <hip/hip_runtime.h>

typedef __bf16 bf16;
typedef __attribute__((ext_vector_type(16))) __bf16 v16bf;
typedef __attribute__((ext_vector_type(8)))  __bf16 bf16x8;
typedef __attribute__((ext_vector_type(8)))  float   v8f;
typedef __attribute__((ext_vector_type(4)))  int     v4i;

#define NB   64
#define NS   1024
#define NI   512
#define NH   512
#define NG3  1536
#define RNWG 16      // workgroups in persistent recurrence
#define RCOLS 32     // h columns per WG
#define RGC   96     // gate columns per WG (3*32)

#define WMMA_BF16(A, Bf, C) \
  __builtin_amdgcn_wmma_f32_16x16x32_bf16(false, (A), false, (Bf), (short)0, (C), false, false)

// ---- CDNA5 async global->LDS path (ASYNCcnt); signature probed: (AS1 int4*, AS3 int4*, imm, imm)
#if defined(__has_builtin)
# if __has_builtin(__builtin_amdgcn_global_load_async_to_lds_b128)
#  define HAS_ASYNC 1
# endif
#endif
#ifndef HAS_ASYNC
# define HAS_ASYNC 0
#endif

#define AS1P __attribute__((address_space(1)))
#define AS3P __attribute__((address_space(3)))

#if HAS_ASYNC
static __device__ __forceinline__ void async_cp16(const bf16* g, bf16* l) {
  __builtin_amdgcn_global_load_async_to_lds_b128((AS1P v4i*)g, (AS3P v4i*)l, 0, 0);
}
# if __has_builtin(__builtin_amdgcn_s_wait_asynccnt)
#  define WAIT_ASYNC() __builtin_amdgcn_s_wait_asynccnt(0)
# else
#  define WAIT_ASYNC() asm volatile("s_wait_asynccnt 0" ::: "memory")
# endif
#endif

// ---- WMMA operand fragment builders (wave32 layouts per CDNA5 ISA 7.12.2) ----
static __device__ __forceinline__ v16bf make_afrag(const bf16* __restrict__ base, int lda) {
  const int lane = threadIdx.x & 31;
  const int row  = lane & 15;
  const int klo  = (lane >> 4) << 3;
  const bf16* p = base + row * lda + klo;
  bf16x8 lo = *(const bf16x8*)(p);
  bf16x8 hi = *(const bf16x8*)(p + 16);
  v16bf f;
#pragma unroll
  for (int e = 0; e < 8; ++e) { f[e] = lo[e]; f[e + 8] = hi[e]; }
  return f;
}
static __device__ __forceinline__ v16bf make_bfrag(const bf16* __restrict__ baseT, int ldt) {
  const int lane = threadIdx.x & 31;
  const int n    = lane & 15;
  const int klo  = (lane >> 4) << 3;
  const bf16* p = baseT + n * ldt + klo;
  bf16x8 lo = *(const bf16x8*)(p);
  bf16x8 hi = *(const bf16x8*)(p + 16);
  v16bf f;
#pragma unroll
  for (int e = 0; e < 8; ++e) { f[e] = lo[e]; f[e + 8] = hi[e]; }
  return f;
}

// ---- 32x32-tile transpose: W[mat][k][h] f32 -> WT[ld][g*512+h][k] bf16 ----
__global__ __launch_bounds__(256) void transpose_w(const float* __restrict__ W,
                                                   bf16* __restrict__ WT, int K) {
  __shared__ float t[32][33];
  const int mat = blockIdx.z;            // 0..11 = ld*3+g
  const int k0 = blockIdx.x << 5;
  const int h0 = blockIdx.y << 5;
  const int tx = threadIdx.x & 31, ty = threadIdx.x >> 5;
  const float* src = W + (long)mat * K * NH;
#pragma unroll
  for (int r = ty; r < 32; r += 8) t[r][tx] = src[(long)(k0 + r) * NH + h0 + tx];
  __syncthreads();
  const int ld = mat / 3, g = mat % 3;
  bf16* dst = WT + ((long)ld * NG3 + (g << 9) + h0) * K + k0;
#pragma unroll
  for (int r = ty; r < 32; r += 8) dst[(long)r * K + tx] = (bf16)t[tx][r];
}

// ---- xs = A(bf16) @ W^T(bf16,[NG3][K]) + bias ; out f32 [Srun,3,NB,NH] ----
// Double-buffered 64x128 tile, 8 waves, 4 WMMA/wave per K-step. Srun = 1<<sshift.
__global__ __launch_bounds__(256) void gemm_xs(
    const bf16* __restrict__ A, long strideA,
    const bf16* __restrict__ BT,       // pre-transposed weights, rows n=g*512+h, cols k
    const float* __restrict__ bias,    // [NG3]
    float* __restrict__ out,
    int K, int sshift)
{
  __shared__ __align__(16) bf16 As[2][64 * 32];
  __shared__ __align__(16) bf16 BTs[2][128 * 32];
  const int tid  = threadIdx.x;
  const int wave = tid >> 5, lane = tid & 31;
  const int m0 = blockIdx.x * 64;
  const int n0 = blockIdx.y * 128;
  const int mw = (wave & 3) << 4;     // wave row offset (4 waves cover M=64)
  const int nb = (wave >> 2) << 6;    // wave col offset (0 or 64; 4 tiles each)
  const int nk = K >> 5;

  const int ar = tid >> 2, ac = (tid & 3) << 3;            // A: 1x16B per thread
  const int bl0 = tid, bl1 = tid + 256;                    // B: 2x16B per thread

  v8f acc[4]; acc[0] = v8f{}; acc[1] = v8f{}; acc[2] = v8f{}; acc[3] = v8f{};

#if HAS_ASYNC
  { // prologue tile 0 via async DMA to LDS
    async_cp16(A + (long)(m0 + ar) * strideA + ac, &As[0][(ar << 5) + ac]);
    async_cp16(BT + (long)(n0 + (bl0 >> 2)) * K + ((bl0 & 3) << 3), &BTs[0][((bl0 >> 2) << 5) + ((bl0 & 3) << 3)]);
    async_cp16(BT + (long)(n0 + (bl1 >> 2)) * K + ((bl1 & 3) << 3), &BTs[0][((bl1 >> 2) << 5) + ((bl1 & 3) << 3)]);
    WAIT_ASYNC();
    __syncthreads();
  }
  for (int kt = 0; kt < nk; ++kt) {
    const int cur = kt & 1, nxt = cur ^ 1;
    if (kt + 1 < nk) { // kick off next tile's DMA before the math
      const int k1 = (kt + 1) << 5;
      async_cp16(A + (long)(m0 + ar) * strideA + k1 + ac, &As[nxt][(ar << 5) + ac]);
      async_cp16(BT + (long)(n0 + (bl0 >> 2)) * K + k1 + ((bl0 & 3) << 3), &BTs[nxt][((bl0 >> 2) << 5) + ((bl0 & 3) << 3)]);
      async_cp16(BT + (long)(n0 + (bl1 >> 2)) * K + k1 + ((bl1 & 3) << 3), &BTs[nxt][((bl1 >> 2) << 5) + ((bl1 & 3) << 3)]);
    }
    v16bf af = make_afrag(&As[cur][mw << 5], 32);
#pragma unroll
    for (int j = 0; j < 4; ++j) {
      v16bf bfj = make_bfrag(&BTs[cur][(nb + (j << 4)) << 5], 32);
      acc[j] = WMMA_BF16(af, bfj, acc[j]);
    }
    WAIT_ASYNC();
    __syncthreads();
  }
#else
  { // prologue tile 0 (register staged)
    bf16x8 ra  = *(const bf16x8*)(A + (long)(m0 + ar) * strideA + ac);
    bf16x8 rb0 = *(const bf16x8*)(BT + (long)(n0 + (bl0 >> 2)) * K + ((bl0 & 3) << 3));
    bf16x8 rb1 = *(const bf16x8*)(BT + (long)(n0 + (bl1 >> 2)) * K + ((bl1 & 3) << 3));
    *(bf16x8*)(&As[0][(ar << 5) + ac]) = ra;
    *(bf16x8*)(&BTs[0][((bl0 >> 2) << 5) + ((bl0 & 3) << 3)]) = rb0;
    *(bf16x8*)(&BTs[0][((bl1 >> 2) << 5) + ((bl1 & 3) << 3)]) = rb1;
    __syncthreads();
  }
  for (int kt = 0; kt < nk; ++kt) {
    const int cur = kt & 1, nxt = cur ^ 1;
    const bool pf = (kt + 1 < nk);
    bf16x8 ra, rb0, rb1;
    if (pf) { // issue next tile's global loads before the math
      const int k1 = (kt + 1) << 5;
      ra  = *(const bf16x8*)(A + (long)(m0 + ar) * strideA + k1 + ac);
      rb0 = *(const bf16x8*)(BT + (long)(n0 + (bl0 >> 2)) * K + k1 + ((bl0 & 3) << 3));
      rb1 = *(const bf16x8*)(BT + (long)(n0 + (bl1 >> 2)) * K + k1 + ((bl1 & 3) << 3));
    }
    v16bf af = make_afrag(&As[cur][mw << 5], 32);
#pragma unroll
    for (int j = 0; j < 4; ++j) {
      v16bf bfj = make_bfrag(&BTs[cur][(nb + (j << 4)) << 5], 32);
      acc[j] = WMMA_BF16(af, bfj, acc[j]);
    }
    if (pf) {
      *(bf16x8*)(&As[nxt][(ar << 5) + ac]) = ra;
      *(bf16x8*)(&BTs[nxt][((bl0 >> 2) << 5) + ((bl0 & 3) << 3)]) = rb0;
      *(bf16x8*)(&BTs[nxt][((bl1 >> 2) << 5) + ((bl1 & 3) << 3)]) = rb1;
    }
    __syncthreads();
  }
#endif

  const int half = (lane >> 4) << 3;
  const int nl   = lane & 15;
  const int smask = (1 << sshift) - 1;
#pragma unroll
  for (int j = 0; j < 4; ++j) {
#pragma unroll
    for (int r = 0; r < 8; ++r) {
      int m = m0 + mw + half + r;
      int s = m & smask, b = m >> sshift;
      int n = n0 + nb + (j << 4) + nl;
      int g = n >> 9, h = n & 511;
      out[(((long)s * 3 + g) * NB + b) * NH + h] = acc[j][r] + bias[n];
    }
  }
}

// ---- persistent GRU recurrence: 16 resident WGs, atomic grid barrier each step ----
static __device__ __forceinline__ void gridsync(unsigned* barp, unsigned* phase) {
  __threadfence();
  __syncthreads();
  if (threadIdx.x == 0) {
    atomicAdd(barp, 1u);
    unsigned tgt = ++(*phase) * (unsigned)RNWG;
    while (atomicAdd(barp, 0u) < tgt) __builtin_amdgcn_s_sleep(1);
    __threadfence();
  }
  __syncthreads();
}

__global__ __launch_bounds__(256) void gru_recur(
    const float* __restrict__ xs,     // [S,3,NB,NH] (bias folded in)
    const bf16*  __restrict__ WhT,    // transposed [NG3][NH] for this (layer, dir): row n=g*512+h, col k
    bf16* __restrict__ hpp,           // [2,NB,NH] ping-pong bf16 state
    bf16* __restrict__ ys,            // [NB,S,NH] or null
    float* __restrict__ hfinal,       // [NB,NH] or null
    unsigned* __restrict__ barp,
    int S)
{
  extern __shared__ char smem[];
  bf16*  WhL   = (bf16*)smem;                                                     // [96][512]  96KB
  float* gates = (float*)(smem + (size_t)RGC * NH * sizeof(bf16));                // [3][64][32] 24KB
  float* hprev = (float*)(smem + (size_t)RGC * NH * sizeof(bf16)
                               + (size_t)3 * NB * RCOLS * sizeof(float));         // [64][32]    8KB
  const int tid  = threadIdx.x;
  const int wave = tid >> 5, lane = tid & 31;
  const int colbase = blockIdx.x * RCOLS;

  // preload this WG's Wh columns (already K-contiguous in WhT): vector copies
  for (int idx = tid; idx < RGC * (NH >> 3); idx += 256) {
    int c  = idx >> 6;              // row (gate col) 0..95
    int kc = (idx & 63) << 3;       // 8-elem chunk
    int g = c >> 5, j = c & 31;
    *(bf16x8*)(&WhL[(c << 9) + kc]) =
        *(const bf16x8*)(WhT + ((long)(g << 9) + colbase + j) * NH + kc);
  }
  // h0 = 0
  for (int idx = tid; idx < NB * RCOLS; idx += 256) {
    hprev[idx] = 0.f;
    int b = idx >> 5, j = idx & 31;
    hpp[(b << 9) + colbase + j] = (bf16)0.f;
  }
  unsigned phase = 0;
  gridsync(barp, &phase);

  const int mw = (wave & 3) << 4;     // 4 waves cover M=64
  const int nb = (wave >> 2) * 48;    // 2 wave groups cover 96 gate cols (3 tiles each)
  const int pb = (tid << 3) >> 5;     // this thread's batch row (for prefetch)
  const int pj = (tid << 3) & 31;
  for (int s = 0; s < S; ++s) {
    const bf16* hb = hpp + ((s & 1) ? (NB * NH) : 0);
    v8f a0 = {}, a1 = {}, a2 = {};
    for (int k0 = 0; k0 < NH; k0 += 32) {
      v16bf af = make_afrag(hb + ((long)mw << 9) + k0, NH);       // h from L2
      v16bf b0 = make_bfrag(&WhL[((nb +  0) << 9) + k0], NH);     // Wh from LDS
      v16bf b1 = make_bfrag(&WhL[((nb + 16) << 9) + k0], NH);
      v16bf b2 = make_bfrag(&WhL[((nb + 32) << 9) + k0], NH);
      a0 = WMMA_BF16(af, b0, a0);
      a1 = WMMA_BF16(af, b1, a1);
      a2 = WMMA_BF16(af, b2, a2);
    }
    const int half = (lane >> 4) << 3;
    const int nl   = lane & 15;
#pragma unroll
    for (int r = 0; r < 8; ++r) {
      int b  = mw + half + r;
      int c0 = nb + nl, c1 = nb + 16 + nl, c2 = nb + 32 + nl;
      gates[((c0 >> 5) << 11) + (b << 5) + (c0 & 31)] = a0[r];
      gates[((c1 >> 5) << 11) + (b << 5) + (c1 & 31)] = a1[r];
      gates[((c2 >> 5) << 11) + (b << 5) + (c2 & 31)] = a2[r];
    }
    __syncthreads();
    const float* xsS = xs + (long)s * (3 * NB * NH);
    bf16* hnext = hpp + ((s & 1) ? 0 : (NB * NH));
#pragma unroll
    for (int i = 0; i < 8; ++i) {
      int idx = (tid << 3) + i;
      int b = idx >> 5, j = idx & 31;
      int col = colbase + j;
      float x0 = xsS[((0 * NB + b) << 9) + col];
      float x1 = xsS[((1 * NB + b) << 9) + col];
      float x2 = xsS[((2 * NB + b) << 9) + col];
      float g0 = gates[(0 << 11) + (b << 5) + j];
      float g1 = gates[(1 << 11) + (b << 5) + j];
      float g2 = gates[(2 << 11) + (b << 5) + j];
      float rg = 1.f / (1.f + __expf(-(x0 + g0)));
      float zg = 1.f / (1.f + __expf(-(x1 + g1)));
      float ng = tanhf(x2 + rg * g2);
      float hp = hprev[(b << 5) + j];
      float hn = (1.f - zg) * ng + zg * hp;
      hprev[(b << 5) + j] = hn;
      hnext[(b << 9) + col] = (bf16)hn;
      if (ys) ys[(((long)b * S) + s) * NH + col] = (bf16)hn;
      if (hfinal && s == S - 1) hfinal[(b << 9) + col] = hn;
    }
    if (s + 1 < S) { // pull next step's xs toward the caches while we wait
      const float* xn = xs + (long)(s + 1) * (3 * NB * NH);
      __builtin_prefetch(xn + ((0 * NB + pb) << 9) + colbase + pj, 0, 1);
      __builtin_prefetch(xn + ((1 * NB + pb) << 9) + colbase + pj, 0, 1);
      __builtin_prefetch(xn + ((2 * NB + pb) << 9) + colbase + pj, 0, 1);
    }
    gridsync(barp, &phase);
  }
}

// ---- backward-direction one-step cell with h0==0: h = (1-sigmoid(xg1))*tanh(xg2) ----
__global__ void gru_cell_h0(const float* __restrict__ xg, float* __restrict__ hout,
                            bf16* __restrict__ houtb) {
  int i = blockIdx.x * blockDim.x + threadIdx.x;
  if (i >= NB * NH) return;
  float z = 1.f / (1.f + __expf(-xg[1 * NB * NH + i]));
  float n = tanhf(xg[2 * NB * NH + i]);
  float h = (1.f - z) * n;
  hout[i] = h;
  if (houtb) houtb[i] = (bf16)h;
}

__global__ void fc_out(const float* __restrict__ hf, const float* __restrict__ hb,
                       const float* __restrict__ fcw, const float* __restrict__ fcb,
                       float* __restrict__ out) {
  int i = blockIdx.x * blockDim.x + threadIdx.x;
  if (i >= NB * NH) return;
  int b = i >> 9, n = i & 511;
  float acc = fcb[n];
  const float* hfb = hf + (b << 9);
  const float* hbb = hb + (b << 9);
  for (int k = 0; k < NH; ++k) acc = fmaf(hfb[k], fcw[k * NH + n], acc);
  for (int k = 0; k < NH; ++k) acc = fmaf(hbb[k], fcw[(NH + k) * NH + n], acc);
  out[i] = acc;
}

__global__ void cvt_bf16(const float* __restrict__ in, bf16* __restrict__ out, long n) {
  long i = (((long)blockIdx.x * blockDim.x) + threadIdx.x) << 2;
  if (i + 3 < n) {
    const float4 v = *(const float4*)(in + i);
    out[i] = (bf16)v.x; out[i + 1] = (bf16)v.y; out[i + 2] = (bf16)v.z; out[i + 3] = (bf16)v.w;
  } else {
    for (; i < n; ++i) out[i] = (bf16)in[i];
  }
}

__global__ void bar_init(unsigned* p) { if (threadIdx.x == 0 && blockIdx.x == 0) *p = 0u; }

extern "C" void kernel_launch(void* const* d_in, const int* in_sizes, int n_in,
                              void* d_out, int out_size, void* d_ws, size_t ws_size,
                              hipStream_t stream) {
  (void)in_sizes; (void)n_in; (void)out_size; (void)ws_size;
  const float* x   = (const float*)d_in[0];
  const float* Wi  = (const float*)d_in[1];
  const float* Wh  = (const float*)d_in[2];
  const float* bb  = (const float*)d_in[3];
  const float* fcw = (const float*)d_in[4];
  const float* fcb = (const float*)d_in[5];
  float* out = (float*)d_out;

  char* ws = (char*)d_ws;
  size_t off = 0;
  auto alloc = [&](size_t bytes) { char* p = ws + off; off = (off + bytes + 255) & ~(size_t)255; return p; };
  bf16*  x_bf  = (bf16*)alloc((size_t)NB * NS * NI * 2);          // 64 MB
  bf16*  wiT   = (bf16*)alloc((size_t)4 * NG3 * NI * 2);          // 6 MB  [ld][n][k]
  bf16*  whT   = (bf16*)alloc((size_t)4 * NG3 * NH * 2);          // 6 MB  [ld][n][k]
  float* xs    = (float*)alloc((size_t)NS * 3 * NB * NH * 4);     // 384 MB
  bf16*  ys0   = (bf16*)alloc((size_t)NB * NS * NH * 2);          // 64 MB
  bf16*  hpp   = (bf16*)alloc((size_t)2 * NB * NH * 2);
  float* hf    = (float*)alloc((size_t)NB * NH * 4);
  float* xgb   = (float*)alloc((size_t)3 * NB * NH * 4);
  float* hb0   = (float*)alloc((size_t)NB * NH * 4);
  bf16*  hb0b  = (bf16*)alloc((size_t)NB * NH * 2);
  float* hb1   = (float*)alloc((size_t)NB * NH * 4);
  unsigned* bar = (unsigned*)alloc(256);

  const long nx = (long)NB * NS * NI;
  cvt_bf16<<<dim3((unsigned)((nx / 4 + 255) / 256)), 256, 0, stream>>>(x, x_bf, nx);
  transpose_w<<<dim3(NI / 32, NH / 32, 12), 256, 0, stream>>>(Wi, wiT, NI);
  transpose_w<<<dim3(NH / 32, NH / 32, 12), 256, 0, stream>>>(Wh, whT, NH);

  const size_t tOff = (size_t)NG3 * NH;      // per (layer,dir) transposed-weight offset
  const size_t bOff = (size_t)3 * NH;
  const size_t SMEM = (size_t)RGC * NH * 2 + (size_t)3 * NB * RCOLS * 4 + (size_t)NB * RCOLS * 4; // 128KB

  // forward, layer 0 (idx 0)
  gemm_xs<<<dim3(NB * NS / 64, NG3 / 128), 256, 0, stream>>>(x_bf, (long)NI, wiT + 0 * tOff, bb + 0 * bOff, xs, NI, 10);
  bar_init<<<1, 1, 0, stream>>>(bar);
  gru_recur<<<RNWG, 256, SMEM, stream>>>(xs, whT + 0 * tOff, hpp, ys0, nullptr, bar, NS);
  // forward, layer 1 (idx 2)
  gemm_xs<<<dim3(NB * NS / 64, NG3 / 128), 256, 0, stream>>>(ys0, (long)NH, wiT + 2 * tOff, bb + 2 * bOff, xs, NH, 10);
  bar_init<<<1, 1, 0, stream>>>(bar);
  gru_recur<<<RNWG, 256, SMEM, stream>>>(xs, whT + 2 * tOff, hpp, nullptr, hf, bar, NS);
  // backward direction: exact collapse (only step 0 of each layer reaches the output)
  gemm_xs<<<dim3(1, NG3 / 128), 256, 0, stream>>>(x_bf + (size_t)(NS - 1) * NI, (long)NS * NI,
                                                  wiT + 1 * tOff, bb + 1 * bOff, xgb, NI, 0);
  gru_cell_h0<<<(NB * NH) / 256, 256, 0, stream>>>(xgb, hb0, hb0b);
  gemm_xs<<<dim3(1, NG3 / 128), 256, 0, stream>>>(hb0b, (long)NH, wiT + 3 * tOff, bb + 3 * bOff, xgb, NH, 0);
  gru_cell_h0<<<(NB * NH) / 256, 256, 0, stream>>>(xgb, hb1, nullptr);
  // final FC on concat(h_fwd[S-1], h_bwd)
  fc_out<<<(NB * NH) / 256, 256, 0, stream>>>(hf, hb1, fcw, fcb, out);
}
